// MoEMLP_61443802137313
// MI455X (gfx1250) — matrix operand order repaired and checked by reading, not compiled, not run on hardware
//
#include <hip/hip_runtime.h>
#include <hip/hip_bf16.h>

// MoE MLP for MI455X (gfx1250): routed top-2, bf16 WMMA (v_wmma_f32_16x16x32_bf16),
// per-expert bf16 weight staging, double-buffered global_load_async_to_lds_b128
// staging (ASYNCcnt) with loop-invariant addresses held in VGPRs,
// deterministic 2x-atomic scatter. Workspace ~101 MB.

#define T_TOK 8192   // B*S
#define HDIM  1024
#define DFF   4096
#define NEXP  8
#define TILE_E (128 * 32)   // ushorts per LDS tile buffer

typedef __attribute__((ext_vector_type(16))) __bf16 v16bf;
typedef __attribute__((ext_vector_type(8)))  float  v8f;

union FragU { uint4 q[2]; v16bf v; };

__device__ __forceinline__ unsigned short f2bf(float f) {
  unsigned int u = __float_as_uint(f);
  u += 0x7fffu + ((u >> 16) & 1u);   // round-to-nearest-even
  return (unsigned short)(u >> 16);
}

__device__ __forceinline__ void async_cp16(unsigned lds_addr, unsigned long long gaddr) {
  // 16B global -> LDS, tracked with ASYNCcnt (one per-wave count per instruction)
  asm volatile("global_load_async_to_lds_b128 %0, %1, off"
               :: "v"(lds_addr), "v"(gaddr) : "memory");
}

// ---------------------------------------------------------------- prep
__global__ __launch_bounds__(256) void moe_prep(const float* __restrict__ X,
                                                unsigned short* __restrict__ Xbf,
                                                float* __restrict__ Out,
                                                int* __restrict__ counts) {
  size_t i = (size_t)blockIdx.x * 256 + threadIdx.x;
  Out[i] = 0.0f;
  Xbf[i] = f2bf(X[i]);
  if (i < NEXP) counts[i] = 0;
}

// ------------------------------------------------------- fp32 -> bf16 weights
__global__ __launch_bounds__(256) void moe_cvt(const float* __restrict__ src,
                                               unsigned short* __restrict__ dst) {
  size_t i4 = ((size_t)blockIdx.x * 256 + threadIdx.x) * 4;
  float4 f = *(const float4*)(src + i4);
  uint2 p;
  p.x = (unsigned int)f2bf(f.x) | ((unsigned int)f2bf(f.y) << 16);
  p.y = (unsigned int)f2bf(f.z) | ((unsigned int)f2bf(f.w) << 16);
  *(uint2*)(dst + i4) = p;
}

// ---------------------------------------------------------------- router
__global__ __launch_bounds__(256) void moe_router(const float* __restrict__ X,
                                                  const float* __restrict__ Wr,
                                                  int* __restrict__ tokIdx,
                                                  float* __restrict__ tokW,
                                                  int* __restrict__ counts) {
  __shared__ float sWr[NEXP * HDIM];
  for (int i = threadIdx.x; i < NEXP * HDIM; i += 256) sWr[i] = Wr[i];
  __syncthreads();

  const int wave = threadIdx.x >> 5, lane = threadIdx.x & 31;
  const int t = blockIdx.x * 8 + wave;

  float acc[NEXP];
#pragma unroll
  for (int e = 0; e < NEXP; ++e) acc[e] = 0.0f;
  for (int h = lane; h < HDIM; h += 32) {
    float xv = X[(size_t)t * HDIM + h];
#pragma unroll
    for (int e = 0; e < NEXP; ++e) acc[e] += xv * sWr[e * HDIM + h];
  }
#pragma unroll
  for (int off = 16; off > 0; off >>= 1) {
#pragma unroll
    for (int e = 0; e < NEXP; ++e) acc[e] += __shfl_xor(acc[e], off, 32);
  }

  if (lane == 0) {
    float best = -INFINITY, second = -INFINITY;
    int bi = 0, si = 0;
#pragma unroll
    for (int e = 0; e < NEXP; ++e) {
      float v = acc[e];
      if (v > best)        { second = best; si = bi; best = v; bi = e; }
      else if (v > second) { second = v; si = e; }
    }
    float p0 = 1.0f / (1.0f + expf(second - best));  // softmax over top-2
    float p1 = 1.0f - p0;
    int s0 = atomicAdd(counts + bi, 1);
    tokIdx[bi * T_TOK + s0] = t;  tokW[bi * T_TOK + s0] = p0;
    int s1 = atomicAdd(counts + si, 1);
    tokIdx[si * T_TOK + s1] = t;  tokW[si * T_TOK + s1] = p1;
  }
}

// ---------------------------------------------------------------- GEMM1 + GELU
__global__ __launch_bounds__(256) void moe_gemm1(
    const unsigned short* __restrict__ Xbf,   // [T][H] bf16
    const unsigned short* __restrict__ W1bf,  // [DFF][H] bf16 (expert staged)
    const float* __restrict__ B1,             // [DFF]
    const int* __restrict__ countP,
    const int* __restrict__ tokIdx,           // [T]
    unsigned short* __restrict__ H1) {        // [T][DFF] bf16
  const int cnt = *countP;
  const int m0 = blockIdx.y * 128;
  if (m0 >= cnt) return;
  const int n0 = blockIdx.x * 128;

  __shared__ unsigned short As[2 * TILE_E];
  __shared__ unsigned short Bs[2 * TILE_E];
  __shared__ int sIdx[128];

  const int tid = threadIdx.x;
  if (tid < 128) {
    int s = m0 + tid;
    sIdx[tid] = (s < cnt) ? tokIdx[s] : tokIdx[0];
  }
  __syncthreads();

  // ---- loop-invariant staging addresses, kept in VGPRs (immune to asm clobber)
  const int r = tid >> 2, q = tid & 3;        // this thread's tile slots
  const unsigned ldsA0 = (unsigned)(uintptr_t)&As[r * 32 + q * 8];
  const unsigned ldsA1 = (unsigned)(uintptr_t)&As[(r + 64) * 32 + q * 8];
  const unsigned ldsB0 = (unsigned)(uintptr_t)&Bs[r * 32 + q * 8];
  const unsigned ldsB1 = (unsigned)(uintptr_t)&Bs[(r + 64) * 32 + q * 8];
  const unsigned long long gA0 =
      (unsigned long long)(uintptr_t)(Xbf + (size_t)sIdx[r] * HDIM + q * 8);
  const unsigned long long gA1 =
      (unsigned long long)(uintptr_t)(Xbf + (size_t)sIdx[r + 64] * HDIM + q * 8);
  const unsigned long long gB0 =
      (unsigned long long)(uintptr_t)(W1bf + (size_t)(n0 + r) * HDIM + q * 8);
  const unsigned long long gB1 =
      (unsigned long long)(uintptr_t)(W1bf + (size_t)(n0 + r + 64) * HDIM + q * 8);

  auto issue = [&](int t, int buf) {
    const unsigned long long kb = (unsigned long long)(t * 64);  // 32 bf16 = 64 B
    const unsigned lb = (unsigned)(buf * TILE_E * 2);
    async_cp16(ldsA0 + lb, gA0 + kb);
    async_cp16(ldsA1 + lb, gA1 + kb);
    async_cp16(ldsB0 + lb, gB0 + kb);
    async_cp16(ldsB1 + lb, gB1 + kb);
  };

  const int lane = tid & 31;
  const int wave = tid >> 5;
  const int wm = (wave & 1) * 64;
  const int wn = (wave >> 1) * 32;

  v8f acc[4][2] = {};

  // ISA 16-bit A (16x32): lane<16 -> K{0..7,16..23}, lane>=16 -> K{8..15,24..31}
  const int arow = lane & 15;
  const int akb  = (lane >> 4) * 8;
  // ISA 16-bit B (32x16): lane<16 -> K{0..7,16..23}, lane>=16 -> K{8..15,24..31}; col = lane&15
  const int bcol = lane & 15;
  const int bkb  = (lane >> 4) * 8;

  auto compute = [&](int buf) {
    const unsigned short* Asb = &As[buf * TILE_E];
    const unsigned short* Bsb = &Bs[buf * TILE_E];
    FragU af[4], bfg[2];
#pragma unroll
    for (int mt = 0; mt < 4; ++mt) {
      int row = wm + mt * 16 + arow;
      af[mt].q[0] = *(const uint4*)&Asb[row * 32 + akb];
      af[mt].q[1] = *(const uint4*)&Asb[row * 32 + akb + 16];
    }
#pragma unroll
    for (int nt = 0; nt < 2; ++nt) {
      int col = wn + nt * 16 + bcol;
      bfg[nt].q[0] = *(const uint4*)&Bsb[col * 32 + bkb];
      bfg[nt].q[1] = *(const uint4*)&Bsb[col * 32 + bkb + 16];
    }
#pragma unroll
    for (int mt = 0; mt < 4; ++mt)
#pragma unroll
      for (int nt = 0; nt < 2; ++nt)
        acc[mt][nt] = __builtin_amdgcn_wmma_f32_16x16x32_bf16(
            false, af[mt].v, false, bfg[nt].v, (short)0, acc[mt][nt],
            false, false);
  };

  const int ntiles = HDIM / 32;
  issue(0, 0);
  for (int t = 0; t < ntiles - 1; ++t) {
    const int buf = t & 1;
    issue(t + 1, buf ^ 1);
    asm volatile("s_wait_asynccnt 0x4" ::: "memory");  // tile t's 4 loads done
    __syncthreads();
    compute(buf);
    __syncthreads();   // all reads of buf done before it is refilled
  }
  asm volatile("s_wait_asynccnt 0x0" ::: "memory");
  __syncthreads();
  compute((ntiles - 1) & 1);

  // Epilogue: C/D layout -> VGPR i holds M=i (lanes 0-15) / M=i+8 (lanes 16-31)
  const int rsel = (lane >> 4) * 8;
#pragma unroll
  for (int nt = 0; nt < 2; ++nt) {
    int col = n0 + wn + nt * 16 + (lane & 15);
    float b1v = B1[col];
#pragma unroll
    for (int mt = 0; mt < 4; ++mt) {
#pragma unroll
      for (int i = 0; i < 8; ++i) {
        int row = wm + mt * 16 + rsel + i;
        int slot = m0 + row;
        if (slot < cnt) {
          float v = acc[mt][nt][i] + b1v;
          float g = 0.5f * v * (1.0f + erff(v * 0.70710678118654752f));
          H1[(size_t)slot * DFF + col] = f2bf(g);
        }
      }
    }
  }
}

// ---------------------------------------------------------------- GEMM2 + scatter
__global__ __launch_bounds__(256) void moe_gemm2(
    const unsigned short* __restrict__ H1,    // [T][DFF] bf16
    const unsigned short* __restrict__ W2bf,  // [H][DFF] bf16 (expert staged)
    const float* __restrict__ B2,             // [H]
    const int* __restrict__ countP,
    const int* __restrict__ tokIdx,
    const float* __restrict__ tokW,
    float* __restrict__ Out) {                // [T][H]
  const int cnt = *countP;
  const int m0 = blockIdx.y * 128;
  if (m0 >= cnt) return;
  const int n0 = blockIdx.x * 128;

  __shared__ unsigned short As[2 * TILE_E];
  __shared__ unsigned short Bs[2 * TILE_E];
  __shared__ int   sIdx[128];
  __shared__ float sW[128];

  const int tid = threadIdx.x;
  if (tid < 128) {
    int s = m0 + tid;
    bool ok = s < cnt;
    sIdx[tid] = ok ? tokIdx[s] : 0;
    sW[tid]   = ok ? tokW[s] : 0.0f;
  }
  __syncthreads();

  const int r = tid >> 2, q = tid & 3;
  const unsigned ldsA0 = (unsigned)(uintptr_t)&As[r * 32 + q * 8];
  const unsigned ldsA1 = (unsigned)(uintptr_t)&As[(r + 64) * 32 + q * 8];
  const unsigned ldsB0 = (unsigned)(uintptr_t)&Bs[r * 32 + q * 8];
  const unsigned ldsB1 = (unsigned)(uintptr_t)&Bs[(r + 64) * 32 + q * 8];
  const unsigned long long gA0 =
      (unsigned long long)(uintptr_t)(H1 + (size_t)(m0 + r) * DFF + q * 8);
  const unsigned long long gA1 =
      (unsigned long long)(uintptr_t)(H1 + (size_t)(m0 + r + 64) * DFF + q * 8);
  const unsigned long long gB0 =
      (unsigned long long)(uintptr_t)(W2bf + (size_t)(n0 + r) * DFF + q * 8);
  const unsigned long long gB1 =
      (unsigned long long)(uintptr_t)(W2bf + (size_t)(n0 + r + 64) * DFF + q * 8);

  auto issue = [&](int t, int buf) {
    const unsigned long long kb = (unsigned long long)(t * 64);
    const unsigned lb = (unsigned)(buf * TILE_E * 2);
    async_cp16(ldsA0 + lb, gA0 + kb);
    async_cp16(ldsA1 + lb, gA1 + kb);
    async_cp16(ldsB0 + lb, gB0 + kb);
    async_cp16(ldsB1 + lb, gB1 + kb);
  };

  const int lane = tid & 31;
  const int wave = tid >> 5;
  const int wm = (wave & 1) * 64;
  const int wn = (wave >> 1) * 32;

  v8f acc[4][2] = {};

  const int arow = lane & 15;
  const int akb  = (lane >> 4) * 8;
  const int bcol = lane & 15;
  const int bkb  = (lane >> 4) * 8;

  auto compute = [&](int buf) {
    const unsigned short* Asb = &As[buf * TILE_E];
    const unsigned short* Bsb = &Bs[buf * TILE_E];
    FragU af[4], bfg[2];
#pragma unroll
    for (int mt = 0; mt < 4; ++mt) {
      int row = wm + mt * 16 + arow;
      af[mt].q[0] = *(const uint4*)&Asb[row * 32 + akb];
      af[mt].q[1] = *(const uint4*)&Asb[row * 32 + akb + 16];
    }
#pragma unroll
    for (int nt = 0; nt < 2; ++nt) {
      int col = wn + nt * 16 + bcol;
      bfg[nt].q[0] = *(const uint4*)&Bsb[col * 32 + bkb];
      bfg[nt].q[1] = *(const uint4*)&Bsb[col * 32 + bkb + 16];
    }
#pragma unroll
    for (int mt = 0; mt < 4; ++mt)
#pragma unroll
      for (int nt = 0; nt < 2; ++nt)
        acc[mt][nt] = __builtin_amdgcn_wmma_f32_16x16x32_bf16(
            false, af[mt].v, false, bfg[nt].v, (short)0, acc[mt][nt],
            false, false);
  };

  const int ntiles = DFF / 32;
  issue(0, 0);
  for (int t = 0; t < ntiles - 1; ++t) {
    const int buf = t & 1;
    issue(t + 1, buf ^ 1);
    asm volatile("s_wait_asynccnt 0x4" ::: "memory");
    __syncthreads();
    compute(buf);
    __syncthreads();
  }
  asm volatile("s_wait_asynccnt 0x0" ::: "memory");
  __syncthreads();
  compute((ntiles - 1) & 1);

  const int rsel = (lane >> 4) * 8;
#pragma unroll
  for (int nt = 0; nt < 2; ++nt) {
    int col = n0 + wn + nt * 16 + (lane & 15);
    float b2v = B2[col];
#pragma unroll
    for (int mt = 0; mt < 4; ++mt) {
#pragma unroll
      for (int i = 0; i < 8; ++i) {
        int row = wm + mt * 16 + rsel + i;
        int slot = m0 + row;
        if (slot < cnt) {
          float v = (acc[mt][nt][i] + b2v) * sW[row];
          atomicAdd(&Out[(size_t)sIdx[row] * HDIM + col], v);
        }
      }
    }
  }
}

// ---------------------------------------------------------------- launch
extern "C" void kernel_launch(void* const* d_in, const int* in_sizes, int n_in,
                              void* d_out, int out_size, void* d_ws, size_t ws_size,
                              hipStream_t stream) {
  (void)in_sizes; (void)n_in; (void)out_size; (void)ws_size;
  const float* X  = (const float*)d_in[0];   // [B,S,H]
  const float* Wr = (const float*)d_in[1];   // [E,H]
  const float* W1 = (const float*)d_in[2];   // [E,DFF,H]
  const float* B1 = (const float*)d_in[3];   // [E,DFF]
  const float* W2 = (const float*)d_in[4];   // [E,H,DFF]
  const float* B2 = (const float*)d_in[5];   // [E,H]
  float* Out = (float*)d_out;                // [T,H]

  char* ws = (char*)d_ws;
  int*   counts = (int*)ws;                                        // 8 ints
  int*   tokIdx = (int*)(ws + 256);                                // E*T
  float* tokW   = (float*)(ws + 256 + (size_t)NEXP * T_TOK * 4);   // E*T
  unsigned short* Xbf  = (unsigned short*)(ws + 256 + (size_t)NEXP * T_TOK * 8);
  unsigned short* H1   = Xbf + (size_t)T_TOK * HDIM;               // T*DFF
  unsigned short* W1bf = H1 + (size_t)T_TOK * DFF;                 // DFF*H (per expert)
  unsigned short* W2bf = W1bf + (size_t)DFF * HDIM;                // H*DFF (per expert)

  moe_prep<<<(T_TOK * HDIM) / 256, 256, 0, stream>>>(X, Xbf, Out, counts);
  moe_router<<<T_TOK / 8, 256, 0, stream>>>(X, Wr, tokIdx, tokW, counts);

  const int cvtBlocks = (DFF * HDIM) / 1024;   // 4 elems/thread, 256 threads
  for (int e = 0; e < NEXP; ++e) {
    moe_cvt<<<cvtBlocks, 256, 0, stream>>>(W1 + (size_t)e * DFF * HDIM, W1bf);
    moe_gemm1<<<dim3(DFF / 128, T_TOK / 128), 256, 0, stream>>>(
        Xbf, W1bf, B1 + (size_t)e * DFF,
        counts + e, tokIdx + (size_t)e * T_TOK, H1);
    moe_cvt<<<cvtBlocks, 256, 0, stream>>>(W2 + (size_t)e * HDIM * DFF, W2bf);
    moe_gemm2<<<dim3(HDIM / 128, T_TOK / 128), 256, 0, stream>>>(
        H1, W2bf, B2 + (size_t)e * HDIM,
        counts + e, tokIdx + (size_t)e * T_TOK, tokW + (size_t)e * T_TOK, Out);
  }
}
